// CelegansHH_14499809591306
// MI455X (gfx1250) — 8-bit matrix operand order repaired and checked by reading, compiled, not run on hardware
//
#include <hip/hip_runtime.h>


#define NN     302
#define NPAD   304
#define TT     256
#define KTILES 5
#define AROW   656
#define ABUF   (16 * AROW)
#define DT_    0.05f

typedef __attribute__((ext_vector_type(16))) int   v16i;
typedef __attribute__((ext_vector_type(8)))  float v8f;

__device__ __forceinline__ float fast_rcp(float x) {
#if __has_builtin(__builtin_amdgcn_rcpf)
  return __builtin_amdgcn_rcpf(x);
#else
  return 1.0f / x;
#endif
}

__device__ __forceinline__ unsigned int f32_to_e4m3_sw(float x) {
  unsigned int u    = __float_as_uint(x);
  unsigned int sign = (u >> 24) & 0x80u;
  unsigned int au   = u & 0x7FFFFFFFu;
  float a = __uint_as_float(au);
  if (a >= 448.0f) return sign | 0x7Eu;
  if (au >= 0x3C800000u) {
    unsigned int m   = au & 0x7FFFFFu;
    unsigned int e   = au >> 23;
    unsigned int lsb = (m >> 20) & 1u;
    m += 0x7FFFFu + lsb;
    unsigned int code = ((e - 120u) << 3) + (m >> 20);
    return sign | (code > 0x7Eu ? 0x7Eu : code);
  }
  float d = a * 512.0f;
  unsigned int di = (unsigned int)(d + 0.5f);
  return sign | di;
}

template <bool HI>
__device__ __forceinline__ unsigned int pack2_e4m3(float a, float b,
                                                   unsigned int old) {
#if __has_builtin(__builtin_amdgcn_cvt_pk_fp8_f32)
  return (unsigned int)__builtin_amdgcn_cvt_pk_fp8_f32(a, b, (int)old, HI);
#else
  unsigned int p = (f32_to_e4m3_sw(a) & 0xFFu) | ((f32_to_e4m3_sw(b) & 0xFFu) << 8);
  return HI ? ((old & 0x0000FFFFu) | (p << 16)) : ((old & 0xFFFF0000u) | p);
#endif
}

__device__ __forceinline__ float hh_vtrap10(float x) {
  float r = x * 0.1f;
  if (fabsf(r) < 1e-6f) return 10.0f * (1.0f + 0.5f * r);
  return x * fast_rcp(1.0f - __expf(-r));
}

__device__ __forceinline__ void hh_rates(float V, float* am, float* bm,
                                         float* ah, float* bh,
                                         float* an, float* bn) {
  *am = 0.1f  * hh_vtrap10(V + 40.0f);
  *bm = 4.0f  * __expf((V + 65.0f) * (-1.0f / 18.0f));
  *ah = 0.07f * __expf((V + 65.0f) * (-0.05f));
  *bh = fast_rcp(1.0f + __expf((V + 35.0f) * (-0.1f)));
  *an = 0.01f * hh_vtrap10(V + 55.0f);
  *bn = 0.125f* __expf((V + 65.0f) * (-0.0125f));
}

__global__ void __launch_bounds__(608)
celegans_hh_kernel(const float* __restrict__ I_ext,
                   const float* __restrict__ chem,
                   const float* __restrict__ gap,
                   float* __restrict__ out) {
  __shared__ unsigned char smem[2 * ABUF];

  const int tid  = threadIdx.x;
  const int lane = tid & 31;
  const int w    = tid >> 5;
  const int hlf  = lane >> 4;
  const int l16  = lane & 15;
  const int ncol = w * 16 + l16;
  const int b0   = (int)blockIdx.x * 16;
  const bool live = (ncol < NN);
  const int   ncc   = live ? ncol : 0;
  const float cmask = live ? 1.0f : 0.0f;

  for (int i = tid; i < 2 * ABUF; i += 608) smem[i] = 0;

  v16i bfrag[KTILES];
#pragma unroll
  for (int kt = 0; kt < KTILES; ++kt) {
#pragma unroll
    for (int v = 0; v < 16; ++v) {
      int kin = ((v >> 2) * 32) + (hlf * 16) + ((v & 3) * 4);
      int kg  = kt * 128 + kin;
      float f[4];
#pragma unroll
      for (int j = 0; j < 4; ++j) {
        int  kk      = kg + j;
        bool in_chem = (kk < NN);
        bool in_gap  = (kk >= NPAD) && (kk < NPAD + NN);
        int  krow    = in_chem ? kk : (in_gap ? (kk - NPAD) : 0);
        const float* src = in_chem ? chem : gap;
        float scale  = in_chem ? 0.1f : (in_gap ? 0.05f : 0.0f);
        f[j] = src[(size_t)krow * NN + ncc] * (scale * cmask);
      }
      unsigned int d = pack2_e4m3<false>(f[0], f[1], 0u);
      d = pack2_e4m3<true>(f[2], f[3], d);
      bfrag[kt][v] = (int)d;
    }
  }

  float gdeg = 0.0f;
  for (int k = 0; k < NN; ++k) gdeg += gap[(size_t)k * NN + ncc];
  gdeg *= 0.05f * cmask;

  float am, bm, ah, bh, an, bn;
  hh_rates(-65.0f, &am, &bm, &ah, &bh, &an, &bn);
  float Vst[8], mst[8], hst[8], nst[8];
#pragma unroll
  for (int r = 0; r < 8; ++r) {
    Vst[r] = -65.0f;
    mst[r] = am / (am + bm);
    hst[r] = ah / (ah + bh);
    nst[r] = an / (an + bn);
  }

  const size_t rstride = (size_t)TT * NN;
  size_t gbase = ((size_t)(b0 + 8 * hlf) * TT) * NN + ncc;

#pragma unroll 1
  for (int t = 0; t < TT; ++t) {
    unsigned char* Ac = smem + (t & 1) * ABUF;

#pragma unroll
    for (int r = 0; r < 8; ++r) {
      int row = r + 8 * hlf;
      float V = Vst[r];
      float act = fast_rcp(1.0f + __expf((V + 20.0f) * (-0.2f)));
      unsigned int pk = pack2_e4m3<false>(act, V + 65.0f, 0u);
      Ac[row * AROW + ncol]        = (unsigned char)(pk & 0xFFu);
      Ac[row * AROW + NPAD + ncol] = (unsigned char)((pk >> 8) & 0xFFu);
    }
    __syncthreads();

    float It[8];
#pragma unroll
    for (int r = 0; r < 8; ++r) It[r] = 0.0f;
    if (live) {
#pragma unroll
      for (int r = 0; r < 8; ++r) It[r] = I_ext[gbase + r * rstride];
      if (t + 1 < TT) {
#pragma unroll
        for (int r = 0; r < 8; ++r)
          __builtin_prefetch(&I_ext[gbase + r * rstride + NN], 0, 1);
      }
    }

    v8f acc = {0.f, 0.f, 0.f, 0.f, 0.f, 0.f, 0.f, 0.f};
#pragma unroll
    for (int kt = 0; kt < KTILES; ++kt) {
      v16i a;
#pragma unroll
      for (int v = 0; v < 16; ++v) {
        int kin = ((v >> 2) * 32) + (hlf * 16) + ((v & 3) * 4);
        a[v] = (int)*(const unsigned int*)(Ac + l16 * AROW + kt * 128 + kin);
      }
      acc = __builtin_amdgcn_wmma_f32_16x16x128_fp8_fp8(a, bfrag[kt], (short)0,
                                                        acc, false, false);
    }

#pragma unroll
    for (int r = 0; r < 8; ++r) {
      float V = Vst[r];
      float Isyn = acc[r] - (V + 65.0f) * gdeg;
      float m = mst[r], hg = hst[r], ng = nst[r];
      float Iion = 120.0f * m * m * m * hg * (V - 50.0f)
                 + 36.0f  * ng * ng * ng * ng * (V + 77.0f)
                 + 0.3f   * (V + 54.387f);
      hh_rates(V, &am, &bm, &ah, &bh, &an, &bn);
      float Vn = V + DT_ * (It[r] + Isyn - Iion);
      mst[r] = m  + DT_ * (am * (1.0f - m)  - bm * m);
      hst[r] = hg + DT_ * (ah * (1.0f - hg) - bh * hg);
      nst[r] = ng + DT_ * (an * (1.0f - ng) - bn * ng);
      Vst[r] = Vn;
    }

    if (live) {
#pragma unroll
      for (int r = 0; r < 8; ++r) out[gbase + r * rstride] = Vst[r];
    }
    gbase += NN;
  }
}

extern "C" void kernel_launch(void* const* d_in, const int* in_sizes, int n_in,
                              void* d_out, int out_size, void* d_ws, size_t ws_size,
                              hipStream_t stream) {
  (void)in_sizes; (void)n_in; (void)out_size; (void)d_ws; (void)ws_size;
  const float* I_ext = (const float*)d_in[0];
  const float* chem  = (const float*)d_in[1];
  const float* gap   = (const float*)d_in[2];
  float* out = (float*)d_out;
  celegans_hh_kernel<<<dim3(512 / 16), dim3(608), 0, stream>>>(I_ext, chem, gap, out);
}
